// NeuralRandomForest_50062138802320
// MI455X (gfx1250) — matrix operand order repaired and checked by reading, not compiled, not run on hardware
//
#include <hip/hip_runtime.h>
#include <hip/hip_bf16.h>

// ---------------------------------------------------------------------------
// NeuralRandomForest on MI455X (gfx1250)
//   B=131072, F=128, T=20, DEPTH=5 -> N=31 internal nodes, L=32 leaves, C=2
//   Dominant work: (B x 128) * (128 x 620) GEMM -> v_wmma_f32_16x16x32_f16
//   (f16 inputs, f32 accumulate). x read (64 MB) sets the ~2.9us HBM floor.
// ---------------------------------------------------------------------------

typedef __attribute__((ext_vector_type(16))) _Float16 v16h;
typedef __attribute__((ext_vector_type(8)))  _Float16 v8h;
typedef __attribute__((ext_vector_type(4)))  _Float16 v4h;
typedef __attribute__((ext_vector_type(8)))  float    v8f;

union V16 { v16h v; v8h p[2]; };

#define NRF_B 131072
#define NRF_F 128
#define NRF_T 20
#define NRF_NI 31      // internal nodes
#define NRF_NL 32      // leaves
// packed weights: [t][ntile(2)][kc(4)][lane(32)][h(16)] halfs = 4096 halfs/tree
#define WPACK_HALFS (NRF_T * 4096)                 // 81920 halfs = 163840 B
#define WLCP_OFF_BYTES (WPACK_HALFS * 2)           // float wlcp[T][2][32] after
#define WLCP_FLOATS (NRF_T * 2 * NRF_NL)           // 1280
#define BIAS_OFF_BYTES (WLCP_OFF_BYTES + WLCP_FLOATS * 4)
#define BIASP_FLOATS (NRF_T * NRF_NL)              // 640 (padded, node31 = 0)

// ---------------------------------------------------------------------------
// Prep A: Wm = split_weights * feature_masks, cast to f16, swizzled into the
// contiguous-K B-fragment order used here (K = h + (lane>=16 ? 16 : 0), N = lane%16), node 31
// padded with zeros. One element per thread; 81920 threads total.
// ---------------------------------------------------------------------------
__global__ void nrf_prep_wpack(const float* __restrict__ sw,
                               const float* __restrict__ fm,
                               _Float16* __restrict__ wpack) {
    int idx = blockIdx.x * 256 + threadIdx.x;     // [0, 81920)
    if (idx >= WPACK_HALFS) return;
    int h  = idx & 15;
    int L  = (idx >> 4) & 31;
    int kc = (idx >> 9) & 3;
    int nt = (idx >> 11) & 1;
    int t  = idx >> 12;
    int n  = nt * 16 + (L & 15);
    int K  = kc * 32 + h + ((L >= 16) ? 16 : 0);
    float v = 0.0f;
    if (n < NRF_NI)
        v = sw[(t * NRF_NI + n) * NRF_F + K] * fm[t * NRF_F + K];
    wpack[idx] = (_Float16)v;
}

// ---------------------------------------------------------------------------
// Prep B: wlcp[t][c][l] = softmax(tree_weights)[t] * softmax(leaf_logits[t,l])[c]
// Layout [t][class][leaf] so each eval lane reads 32 contiguous floats.
// ---------------------------------------------------------------------------
__global__ void nrf_prep_wlcp(const float* __restrict__ ll,
                              const float* __restrict__ tw,
                              float* __restrict__ wlcp) {
    int i = blockIdx.x * 256 + threadIdx.x;       // [0, 1280)
    if (i >= WLCP_FLOATS) return;
    int l = i & 31;
    int c = (i >> 5) & 1;
    int t = i >> 6;
    float s = 0.0f;
    for (int j = 0; j < NRF_T; ++j) s += __expf(tw[j]);
    float w = __expf(tw[t]) / s;
    float e0 = __expf(ll[(t * NRF_NL + l) * 2 + 0]);
    float e1 = __expf(ll[(t * NRF_NL + l) * 2 + 1]);
    wlcp[i] = w * (c ? e1 : e0) / (e0 + e1);
}

// ---------------------------------------------------------------------------
// Prep C: bias padded to T x 32 (node 31 = 0) -> branch-free main loop.
// ---------------------------------------------------------------------------
__global__ void nrf_prep_bias(const float* __restrict__ sb,
                              float* __restrict__ biasP) {
    int i = blockIdx.x * 256 + threadIdx.x;       // [0, 640)
    if (i >= BIASP_FLOATS) return;
    int n = i & 31;
    int t = i >> 5;
    biasP[i] = (n < NRF_NI) ? sb[t * NRF_NI + n] : 0.0f;
}

// ---------------------------------------------------------------------------
// Main kernel: 8 waves / block, 16 rows of B per wave, 128 rows per block.
// ---------------------------------------------------------------------------
#define WAVES 8
#define XPAD  136   // 128 + 8 halfs row pad (272B = 17*16 -> aligned, no bank clash)
#define SPAD  36    // 144B row stride: 16B-aligned b128 reads, 16 distinct banks

__global__ __launch_bounds__(256) void nrf_main(
        const float* __restrict__ x,
        const float* __restrict__ biasP,
        const _Float16* __restrict__ wpack,
        const float* __restrict__ wlcp,
        float* __restrict__ out) {
    __shared__ __align__(16) _Float16 xT[WAVES][16][XPAD];  // f16 x tile per wave
    __shared__ __align__(16) float    sP[WAVES][16][SPAD];  // sigmoid probs
    __shared__ __align__(16) float    biasS[BIASP_FLOATS];  // 640
    __shared__ __align__(16) float    wlcpS[WLCP_FLOATS];   // 1280

    const int tid = threadIdx.x;
    const int w   = tid >> 5;
    const int L   = tid & 31;

    for (int i = tid; i < BIASP_FLOATS; i += 256) biasS[i] = biasP[i];
    for (int i = tid; i < WLCP_FLOATS;  i += 256) wlcpS[i] = wlcp[i];

    const int r0 = blockIdx.x * (WAVES * 16) + w * 16;

    // Stage x tile (coalesced float4 loads -> f16 LDS).
    for (int row = 0; row < 16; ++row) {
        const float4 f = *(const float4*)(x + (size_t)(r0 + row) * NRF_F + L * 4);
        v4h h4;
        h4[0] = (_Float16)f.x; h4[1] = (_Float16)f.y;
        h4[2] = (_Float16)f.z; h4[3] = (_Float16)f.w;
        *(v4h*)&xT[w][row][L * 4] = h4;
    }
    __syncthreads();

    // A fragments: contiguous-K order matching B (K = h + (lanes>=16 ? 16 : 0)).
    // Contiguous 8-half runs -> two ds_load_b128 each.
    const int hiK  = (L >= 16) ? 16 : 0;
    const int rowA = L & 15;
    V16 afrag[4];
    #pragma unroll
    for (int kc = 0; kc < 4; ++kc) {
        const int K0 = kc * 32 + hiK;
        afrag[kc].p[0] = *(const v8h*)&xT[w][rowA][K0];
        afrag[kc].p[1] = *(const v8h*)&xT[w][rowA][K0 + 8];
    }

    const int rowE = L & 15;       // row this lane evaluates
    const int cls  = L >> 4;       // class this lane evaluates
    const int rowC = (L >> 4) * 8; // C-matrix M offset (lanes>=16 -> M=v+8)
    const int n0   = L & 15;
    const int n1   = 16 + (L & 15);
    float acc = 0.0f;

    for (int t = 0; t < NRF_T; ++t) {
        if (t + 1 < NRF_T)
            __builtin_prefetch(wpack + (size_t)(t + 1) * 4096, 0, 0);

        // B fragments for this tree (pre-swizzled; L2-resident).
        const v8h* wp = (const v8h*)(wpack + (size_t)t * 4096);
        V16 bf[2][4];
        #pragma unroll
        for (int nt = 0; nt < 2; ++nt)
            #pragma unroll
            for (int kc = 0; kc < 4; ++kc) {
                const int off = (nt * 2048 + kc * 512 + L * 16) >> 3; // v8h units
                bf[nt][kc].p[0] = wp[off];
                bf[nt][kc].p[1] = wp[off + 1];
            }

        v8f c0 = {}; v8f c1 = {};
        #pragma unroll
        for (int kc = 0; kc < 4; ++kc) {
            c0 = __builtin_amdgcn_wmma_f32_16x16x32_f16(
                    false, afrag[kc].v, false, bf[0][kc].v, (short)0, c0, false, false);
            c1 = __builtin_amdgcn_wmma_f32_16x16x32_f16(
                    false, afrag[kc].v, false, bf[1][kc].v, (short)0, c1, false, false);
        }

        // bias + sigmoid, scatter to LDS probs (node 31 is zero-padding;
        // padded bias makes this branch-free).
        const float b0 = biasS[t * 32 + n0];
        const float b1 = biasS[t * 32 + n1];
        #pragma unroll
        for (int v = 0; v < 8; ++v) {
            const int row = v + rowC;
            const float z0 = c0[v] + b0;
            const float z1 = c1[v] + b1;
            sP[w][row][n0] = 1.0f / (1.0f + __expf(-z0));
            sP[w][row][n1] = 1.0f / (1.0f + __expf(-z1));
        }
        asm volatile("s_wait_dscnt 0" ::: "memory");

        // Vectorized LDS reads: this row's 31 probs as 8x b128, this lane's
        // 32 leaf weights (softmax(tree_w)*softmax(leaf_logits) folded) as
        // 8x b128.
        float g[NRF_NL];
        float cur[NRF_NL];
        #pragma unroll
        for (int i = 0; i < 8; ++i) {
            const float4 q = *(const float4*)&sP[w][rowE][4 * i];
            g[4 * i + 0] = q.x; g[4 * i + 1] = q.y;
            g[4 * i + 2] = q.z; g[4 * i + 3] = q.w;
            const float4 u = *(const float4*)&wlcpS[t * 64 + cls * 32 + 4 * i];
            cur[4 * i + 0] = u.x; cur[4 * i + 1] = u.y;
            cur[4 * i + 2] = u.z; cur[4 * i + 3] = u.w;
        }
        // Bottom-up tree evaluation: val[n] = vl + g*(vr - vl).
        #pragma unroll
        for (int n = NRF_NI - 1; n >= 0; --n)
            cur[n] = fmaf(g[n], cur[2 * n + 2] - cur[2 * n + 1], cur[2 * n + 1]);
        acc += cur[0];
        asm volatile("s_wait_dscnt 0" ::: "memory");
    }

    out[(size_t)(r0 + rowE) * 2 + cls] = acc;
}

// ---------------------------------------------------------------------------
// Launch
// ---------------------------------------------------------------------------
extern "C" void kernel_launch(void* const* d_in, const int* in_sizes, int n_in,
                              void* d_out, int out_size, void* d_ws, size_t ws_size,
                              hipStream_t stream) {
    const float* x    = (const float*)d_in[0];  // (B,128)
    const float* sw   = (const float*)d_in[1];  // (T,31,128)
    const float* sb   = (const float*)d_in[2];  // (T,31)
    const float* ll   = (const float*)d_in[3];  // (T,32,2)
    const float* tw   = (const float*)d_in[4];  // (T,)
    const float* fm   = (const float*)d_in[5];  // (T,128)
    float* out        = (float*)d_out;          // (B,2)

    _Float16* wpack = (_Float16*)d_ws;                          // 163840 B
    float*    wlcp  = (float*)((char*)d_ws + WLCP_OFF_BYTES);   // 5120 B
    float*    biasP = (float*)((char*)d_ws + BIAS_OFF_BYTES);   // 2560 B

    nrf_prep_wpack<<<WPACK_HALFS / 256, 256, 0, stream>>>(sw, fm, wpack);
    nrf_prep_wlcp<<<(WLCP_FLOATS + 255) / 256, 256, 0, stream>>>(ll, tw, wlcp);
    nrf_prep_bias<<<(BIASP_FLOATS + 255) / 256, 256, 0, stream>>>(sb, biasP);
    nrf_main<<<NRF_B / (WAVES * 16), 256, 0, stream>>>(x, sb ? biasP : biasP, wpack, wlcp, out);
}
